// RNN_tanh_1589137900337
// MI455X (gfx1250) — compile-verified, matrix operand order checked by reading, not run on hardware
//
#include <hip/hip_runtime.h>
#include <hip/hip_bf16.h>
#include <stdint.h>

typedef unsigned short ushort_t;
typedef __attribute__((ext_vector_type(16))) __bf16 v16bf;
typedef __attribute__((ext_vector_type(8)))  float  v8f;

#define T_STEPS 8192
#define DIM     2048   // D == H == 2048
#define SCAN_WG 32     // persistent workgroups for the recurrence
#define SCAN_THREADS 512
#define ROWS_PER_WG (DIM / SCAN_WG)   // 64 rows of W_hh per WGP, resident in LDS

// ---------- CDNA5 async global->LDS path (guarded; falls back cleanly) ------
#if __has_builtin(__builtin_amdgcn_global_load_async_to_lds_b128)
#define HAVE_ASYNC_LDS 1
typedef int v4i_vs __attribute__((vector_size(16)));   // matches builtin pointee
__device__ __forceinline__ void async_cp16(const void* g, void* l) {
  __builtin_amdgcn_global_load_async_to_lds_b128(
      (__attribute__((address_space(1))) v4i_vs*)(uintptr_t)g,
      (__attribute__((address_space(3))) v4i_vs*)(uint32_t)(uintptr_t)l,
      0, 0);
}
#endif

__device__ __forceinline__ void wait_async_lds() {
#if defined(HAVE_ASYNC_LDS)
# if __has_builtin(__builtin_amdgcn_s_wait_asynccnt)
  __builtin_amdgcn_s_wait_asynccnt(0);
# else
  asm volatile("s_wait_asynccnt 0x0" ::: "memory");
# endif
#endif
}

// ---------- bf16 helpers (RNE) ----------
__device__ __forceinline__ ushort_t f2bf(float f) {
  uint32_t u = __float_as_uint(f);
  u += 0x7FFFu + ((u >> 16) & 1u);
  return (ushort_t)(u >> 16);
}

// ---------- conversion: fp32 -> bf16 (grid-stride) ----------
__global__ void k_f32_to_bf16(const float* __restrict__ in,
                              ushort_t* __restrict__ out, int n) {
  int i = blockIdx.x * blockDim.x + threadIdx.x;
  int stride = gridDim.x * blockDim.x;
  for (; i < n; i += stride) out[i] = f2bf(in[i]);
}

// ---------- prep: fused bias, h0 = 0, zero barrier counters ----------
__global__ void k_prep(const float* __restrict__ b_ih, const float* __restrict__ b_hh,
                       float* __restrict__ bsum, ushort_t* __restrict__ hbuf,
                       unsigned* __restrict__ cnt) {
  int i = blockIdx.x * blockDim.x + threadIdx.x;
  if (i < DIM) {
    bsum[i] = b_ih[i] + b_hh[i];
    hbuf[i] = 0;          // h0 = 0 (buffer 0)
    hbuf[DIM + i] = 0;    // buffer 1
  }
  if (i < T_STEPS) cnt[i] = 0u;
}

// ---------- WMMA fragment load from an LDS tile (row-major, ld = 32 bf16) ----
// CDNA5 16-bit A/B operand layout (ISA 7.12.2):
//   lane<16 : row = lane,    K = {0..7} then {16..23}
//   lane>=16: row = lane-16, K = {8..15} then {24..31}
union Frag { uint4 q[2]; v16bf v; };

__device__ __forceinline__ v16bf lds_frag(const ushort_t* base, int lane) {
  const int r = lane & 15;
  const int hk = (lane >> 4) << 3;   // 0 or 8
  Frag f;
  f.q[0] = *(const uint4*)(base + r * 32 + hk);        // K = hk .. hk+7
  f.q[1] = *(const uint4*)(base + r * 32 + 16 + hk);   // K = 16+hk .. 16+hk+7
  return f.v;
}

// ---------- bf16 WMMA GEMM:  C[M,N] = act( A[M,K] * B[N,K]^T + bias[N] ) ----
// Block tile 128x128, 8 waves (4 along M x 2 along N), wave tile 32x64.
// Ping-pong LDS buffers: tile kt+1 staged (async if available) during WMMA of
// tile kt; single __syncthreads per K-step.
// ACT: 0 = identity, 1 = sigmoid (both fp32 out)
template <int ACT>
__global__ void __launch_bounds__(256)
k_wmma_gemm(const ushort_t* __restrict__ A, const ushort_t* __restrict__ B,
            const float* __restrict__ bias, float* __restrict__ C,
            int M, int N, int K) {
  __shared__ ushort_t sA[2][128 * 32];
  __shared__ ushort_t sB[2][128 * 32];

  const int tid  = threadIdx.x;
  const int lane = tid & 31;
  const int wave = tid >> 5;        // 0..7
  const int wm   = wave & 3;        // wave row   (4 x 32 = 128 M)
  const int wn   = wave >> 2;       // wave col   (2 x 64 = 128 N)
  const int mBase = blockIdx.y * 128;
  const int nBase = blockIdx.x * 128;

  v8f acc[2][4] = {};

  // staging map: 128 rows x 32 cols bf16; 2 threads/row, 16 bf16 per thread
  const int ldRow = tid >> 1;
  const int ldCol = (tid & 1) * 16;
  const size_t gAoff = (size_t)(mBase + ldRow) * K + ldCol;
  const size_t gBoff = (size_t)(nBase + ldRow) * K + ldCol;

  auto stage = [&](int buf, int k0) {
    const uint4* ga = (const uint4*)(A + gAoff + k0);
    const uint4* gb = (const uint4*)(B + gBoff + k0);
    uint4* la = (uint4*)(&sA[buf][ldRow * 32 + ldCol]);
    uint4* lb = (uint4*)(&sB[buf][ldRow * 32 + ldCol]);
#if defined(HAVE_ASYNC_LDS)
    async_cp16(ga,     la);
    async_cp16(ga + 1, la + 1);
    async_cp16(gb,     lb);
    async_cp16(gb + 1, lb + 1);
#else
    la[0] = ga[0]; la[1] = ga[1];
    lb[0] = gb[0]; lb[1] = gb[1];
#endif
  };

  stage(0, 0);
  wait_async_lds();
  __syncthreads();

  const int nk = K / 32;
  for (int kt = 0; kt < nk; ++kt) {
    const int buf = kt & 1;
    if (kt + 1 < nk) stage(buf ^ 1, (kt + 1) * 32);
    if (kt + 2 < nk) {   // warm L2 two tiles ahead (global_prefetch_b8)
      __builtin_prefetch((const void*)(A + gAoff + (size_t)(kt + 2) * 32), 0, 0);
      __builtin_prefetch((const void*)(B + gBoff + (size_t)(kt + 2) * 32), 0, 0);
    }

    v16bf af[2], bfr[4];
#pragma unroll
    for (int mt = 0; mt < 2; ++mt)
      af[mt] = lds_frag(&sA[buf][(wm * 32 + mt * 16) * 32], lane);
#pragma unroll
    for (int nt = 0; nt < 4; ++nt)
      bfr[nt] = lds_frag(&sB[buf][(wn * 64 + nt * 16) * 32], lane);

#pragma unroll
    for (int mt = 0; mt < 2; ++mt)
#pragma unroll
      for (int nt = 0; nt < 4; ++nt)
        acc[mt][nt] = __builtin_amdgcn_wmma_f32_16x16x32_bf16(
            false, af[mt], false, bfr[nt], (short)0, acc[mt][nt], false, false);

    if (kt + 1 < nk) wait_async_lds();
    __syncthreads();
  }

  // epilogue: C/D layout -> VGPR j: lanes 0-15 = (M=j, N=lane),
  //                                 lanes 16-31 = (M=8+j, N=lane-16)
  const int cn = lane & 15;
  const int mh = (lane >> 4) * 8;
#pragma unroll
  for (int mt = 0; mt < 2; ++mt) {
#pragma unroll
    for (int nt = 0; nt < 4; ++nt) {
      const int col = nBase + wn * 64 + nt * 16 + cn;
      const float bv = bias[col];
#pragma unroll
      for (int j = 0; j < 8; ++j) {
        const int row = mBase + wm * 32 + mt * 16 + mh + j;
        float v = acc[mt][nt][j] + bv;
        if (ACT == 1) v = 1.f / (1.f + __expf(-v));
        C[(size_t)row * N + col] = v;
      }
    }
  }
}

// ---------- persistent recurrence: h_t = tanh(pre_t + W_hh h_{t-1}) ---------
// 32 workgroups x 512 threads; each keeps 64 rows of bf16 W_hh (256 KB)
// resident in LDS (<= 320 KB/WGP). h (4 KB bf16) is exchanged through global
// memory (double-buffered) with a per-step atomic counter barrier.
__global__ void __launch_bounds__(SCAN_THREADS)
k_rnn_scan(const ushort_t* __restrict__ Whh, const float* __restrict__ pre,
           ushort_t* __restrict__ hbuf, ushort_t* __restrict__ hs,
           unsigned* __restrict__ cnt, int nwg) {
  extern __shared__ ushort_t sW[];   // ROWS_PER_WG * DIM bf16 = 256 KB
  __shared__ ushort_t sH[DIM];       // staged h (4 KB)

  const int tid = threadIdx.x;
  const int wg  = blockIdx.x;
  const int row = tid >> 3;          // 0..63  (local W_hh row)
  const int seg = tid & 7;           // 8 threads cooperate per row
  const int rowG = wg * ROWS_PER_WG + row;

  {   // pin this WG's W_hh shard into LDS
    const uint4* gw = (const uint4*)(Whh + (size_t)wg * ROWS_PER_WG * DIM);
    uint4* lw = (uint4*)sW;
    for (int i = tid; i < (ROWS_PER_WG * DIM) / 8; i += SCAN_THREADS) lw[i] = gw[i];
  }
  __syncthreads();

  const uint32_t* wr = (const uint32_t*)(sW + row * DIM + seg * 256);

  for (int t = 0; t < T_STEPS; ++t) {
    // stage full h into LDS: 512 threads x 8 B = 4 KB
    ((uint2*)sH)[tid] = ((const uint2*)(hbuf + (t & 1) * DIM))[tid];
    if (seg == 0 && t + 1 < T_STEPS)
      __builtin_prefetch((const void*)&pre[(size_t)(t + 1) * DIM + rowG], 0, 0);
    __syncthreads();

    const uint32_t* hr = (const uint32_t*)(sH + seg * 256);
    float y = 0.f;
#pragma unroll 8
    for (int k = 0; k < 128; ++k) {
      uint32_t w = wr[k], h = hr[k];
      y = fmaf(__uint_as_float(w << 16),         __uint_as_float(h << 16),         y);
      y = fmaf(__uint_as_float(w & 0xFFFF0000u), __uint_as_float(h & 0xFFFF0000u), y);
    }
    // reduce across the 8 cooperating lanes (8-aligned groups within wave32)
    y += __shfl_xor(y, 1);
    y += __shfl_xor(y, 2);
    y += __shfl_xor(y, 4);

    if (seg == 0) {
      float hv = tanhf(pre[(size_t)t * DIM + rowG] + y);
      ushort_t hb = f2bf(hv);
      hbuf[((t + 1) & 1) * DIM + rowG] = hb;   // next step's h
      hs[(size_t)t * DIM + rowG] = hb;         // history for output GEMM
    }
    __threadfence();
    __syncthreads();
    if (tid == 0) {
      __hip_atomic_fetch_add(&cnt[t], 1u, __ATOMIC_RELEASE, __HIP_MEMORY_SCOPE_AGENT);
      while (__hip_atomic_load(&cnt[t], __ATOMIC_ACQUIRE, __HIP_MEMORY_SCOPE_AGENT)
             < (unsigned)nwg) {
        __builtin_amdgcn_s_sleep(8);
      }
    }
    __syncthreads();   // also protects sH reuse next iteration
  }
}

// ---------------------------- launcher --------------------------------------
extern "C" void kernel_launch(void* const* d_in, const int* in_sizes, int n_in,
                              void* d_out, int out_size, void* d_ws, size_t ws_size,
                              hipStream_t stream) {
  (void)in_sizes; (void)n_in; (void)out_size; (void)ws_size;

  const float* x    = (const float*)d_in[0];   // [1,T,D]
  const float* W_ih = (const float*)d_in[1];   // [H,D]
  const float* W_hh = (const float*)d_in[2];   // [H,H]
  const float* b_ih = (const float*)d_in[3];   // [H]
  const float* b_hh = (const float*)d_in[4];   // [H]
  const float* W_o  = (const float*)d_in[5];   // [D,H]
  const float* b_o  = (const float*)d_in[6];   // [D]
  float* out = (float*)d_out;                  // [1,T,D] fp32

  char* ws = (char*)d_ws;
  const size_t MB = 1ull << 20;
  float*    pre  = (float*)(ws);                       // 64 MB  [T,H] fp32
  ushort_t* xb   = (ushort_t*)(ws + 64 * MB);          // 32 MB  x  bf16
  ushort_t* hsb  = (ushort_t*)(ws + 96 * MB);          // 32 MB  hs bf16
  ushort_t* wih  = (ushort_t*)(ws + 128 * MB);         //  8 MB
  ushort_t* whh  = (ushort_t*)(ws + 136 * MB);         //  8 MB
  ushort_t* wo   = (ushort_t*)(ws + 144 * MB);         //  8 MB
  float*    bsum = (float*)(ws + 152 * MB);            //  8 KB
  ushort_t* hbuf = (ushort_t*)(ws + 152 * MB + 8192);  //  8 KB (double-buffered h)
  unsigned* cnt  = (unsigned*)(ws + 152 * MB + 16384); // 32 KB barrier counters

  // 1) precision conversion (weights once per launch; x for GEMM A-operand)
  k_f32_to_bf16<<<4096, 256, 0, stream>>>(x,    xb,  T_STEPS * DIM);
  k_f32_to_bf16<<<2048, 256, 0, stream>>>(W_ih, wih, DIM * DIM);
  k_f32_to_bf16<<<2048, 256, 0, stream>>>(W_hh, whh, DIM * DIM);
  k_f32_to_bf16<<<2048, 256, 0, stream>>>(W_o,  wo,  DIM * DIM);
  k_prep<<<T_STEPS / 256, 256, 0, stream>>>(b_ih, b_hh, bsum, hbuf, cnt);

  // 2) pre = x @ W_ih^T + (b_ih + b_hh)   (WMMA bf16 -> fp32)
  dim3 g1(DIM / 128, T_STEPS / 128);
  k_wmma_gemm<0><<<g1, 256, 0, stream>>>(xb, wih, bsum, pre, T_STEPS, DIM, DIM);

  // 3) sequential scan with LDS-resident W_hh shards
  k_rnn_scan<<<SCAN_WG, SCAN_THREADS, ROWS_PER_WG * DIM * sizeof(ushort_t), stream>>>(
      whh, pre, hbuf, hsb, cnt, SCAN_WG);

  // 4) out = sigmoid(hs @ W_o^T + b_o)    (WMMA bf16 -> fp32)
  k_wmma_gemm<1><<<g1, 256, 0, stream>>>(hsb, wo, b_o, out, T_STEPS, DIM, DIM);
}
